// OuterProductMean_53833120088455
// MI455X (gfx1250) — matrix operand order repaired and checked by reading, not compiled, not run on hardware
//
#include <hip/hip_runtime.h>

typedef __attribute__((ext_vector_type(16))) __bf16 v16bf;
typedef __attribute__((ext_vector_type(8)))  float  v8f;

#define C_IN    32
#define CZ      128
#define S_DEPTH 256
#define NRES    384
#define KSTEPS  8          // S / 32
#define FRAG    512        // elems per 16x16x32 bf16 fragment (32 lanes * 16)

#define WMMA_BF16(A, B, C) \
    __builtin_amdgcn_wmma_f32_16x16x32_bf16(false, (A), false, (B), (short)0, (C), false, false)

// ---------------------------------------------------------------------------
// Stage 1: LayerNorm + W1/W2 projections, emitted directly in WMMA fragment
// layouts (bf16).  One wave32 per (s, i); lane == channel c.
//
// A-fragment (16x32 bf16, M=c_local, K=s_local) element (lane L, elem e):
//   M = L&15,  K = (e<8?0:16) + (L>>4)*8 + (e&7)
// B-fragment (32x16 bf16, K=s_local, N=d_local) element (lane L, elem e):
//   N = L&15,  K = (e<8?0:16) + (L>>4)*8 + (e&7)
// ---------------------------------------------------------------------------
__global__ __launch_bounds__(256)
void opm_prep_ab(const float* __restrict__ m, const float* __restrict__ gamma,
                 const float* __restrict__ beta, const float* __restrict__ W1,
                 const float* __restrict__ W2,
                 __bf16* __restrict__ a_frag, __bf16* __restrict__ b_frag) {
    const int lane = threadIdx.x & 31;
    const int wave = (blockIdx.x * blockDim.x + threadIdx.x) >> 5;   // 0 .. S*N-1
    const int s = wave / NRES;
    const int i = wave % NRES;

    // LayerNorm over the 32 channels (one per lane)
    float x = m[((size_t)s * NRES + i) * C_IN + lane];
    float sum = x;
    #pragma unroll
    for (int off = 16; off >= 1; off >>= 1) sum += __shfl_xor(sum, off, 32);
    const float mu = sum * (1.0f / 32.0f);
    const float d0 = x - mu;
    float vs = d0 * d0;
    #pragma unroll
    for (int off = 16; off >= 1; off >>= 1) vs += __shfl_xor(vs, off, 32);
    const float rstd = rsqrtf(vs * (1.0f / 32.0f) + 1e-5f);
    const float mn = d0 * rstd * gamma[lane] + beta[lane];

    // Projections: lane d accumulates sum_c mn[c] * W[c][d]
    float av = 0.0f, bv = 0.0f;
    #pragma unroll
    for (int c = 0; c < 32; ++c) {
        const float v = __shfl(mn, c, 32);
        av += v * W1[c * C_IN + lane];
        bv += v * W2[c * C_IN + lane];
    }

    const int sl = s & 31;            // K within this fragment's k-step
    const int k  = s >> 5;            // k-step 0..7
    const int d  = lane;              // output channel
    const int tile = d >> 4;          // which 16-wide c/d tile

    // scatter into A-fragment layout (keyed by i)
    {
        const int halfA = (sl >> 3) & 1;
        const int eA    = (sl & 7) + (((sl >> 4) & 1) << 3);
        const int laneA = (d & 15) + (halfA << 4);
        a_frag[((((size_t)i * 2 + tile) * KSTEPS + k) << 9) + laneA * 16 + eA] = (__bf16)av;
    }
    // scatter into B-fragment layout (keyed by j==i)
    {
        const int laneB = (d & 15) + (((sl >> 3) & 1) << 4);
        const int eB    = (sl & 7) + (((sl >> 4) & 1) << 3);
        b_frag[((((size_t)i * 2 + tile) * KSTEPS + k) << 9) + laneB * 16 + eB] = (__bf16)bv;
    }
}

// ---------------------------------------------------------------------------
// Stage 1b: reformat W3 [1024 x 128] fp32 -> bf16 B-fragments [8 ntile][32 ks]
// ---------------------------------------------------------------------------
__global__ __launch_bounds__(256)
void opm_prep_w3(const float* __restrict__ W3, __bf16* __restrict__ w3f) {
    const int lane = threadIdx.x & 31;
    const int frag = (blockIdx.x * blockDim.x + threadIdx.x) >> 5;   // 0..255
    const int nt = frag >> 5;            // output 16-col tile 0..7
    const int ks = frag & 31;            // k-step 0..31 (K = 1024)
    const int n  = nt * 16 + (lane & 15);
    const int kb = ks * 32 + ((lane >> 4) << 3);
    #pragma unroll
    for (int e = 0; e < 16; ++e)
        w3f[(size_t)frag * FRAG + lane * 16 + e] = (__bf16)W3[(size_t)(kb + (e & 7) + ((e >> 3) << 4)) * CZ + n];
}

// ---------------------------------------------------------------------------
// Stage 2+3 fused.  Workgroup = 16 waves = 8(i) x 4(j) pair tile (32 pairs).
//  - Per k-step, 24 fragments (16 A + 8 B, 24 KB) are staged to LDS once via
//    double-buffered GLOBAL_LOAD_ASYNC_TO_LDS_B128 (s_wait_asynccnt sync).
//  - Each wave computes 2 pairs (1 i, 2 j) = 8 WMMAs / k-step from LDS frags.
//  - O (scaled 1/S) is scattered to LDS as stage-3 A-fragments: M = 32 pairs
//    = 2 m-tiles; then [32 x 1024] @ W3[1024 x 128] = 2x8 = 16 output tiles,
//    exactly one per wave over full K (no partial reduction needed).
// ---------------------------------------------------------------------------
__global__ __launch_bounds__(512)
void opm_main(const __bf16* __restrict__ a_frag, const __bf16* __restrict__ b_frag,
              const __bf16* __restrict__ w3f, const float* __restrict__ b3,
              float* __restrict__ z) {
    __shared__ __align__(16) __bf16 stagebuf[2][24 * FRAG];   // 2 x 24 KB staging
    __shared__ __align__(16) __bf16 ofrag[2][32 * FRAG];      // 2 m-tiles x 32 KB

    const int lane    = threadIdx.x & 31;
    const int w       = threadIdx.x >> 5;       // wave 0..15
    const int lane_hi = lane >> 4;
    const int i0 = blockIdx.x * 8;
    const int j0 = blockIdx.y * 4;

    const int il  = w >> 1;                     // i_local 0..7
    const int jl0 = (w & 1) * 2;                // first j_local of this wave

    // ---- async stager: 1536 x 16B chunks (24 frags), 3 per thread ----
    auto stage_k = [&](int k, int buf) {
        const unsigned lbase = (unsigned)(uintptr_t)(&stagebuf[buf][0]);
        #pragma unroll
        for (int q = 0; q < 3; ++q) {
            const int cidx = (int)threadIdx.x + q * 512;   // 0..1535
            const int frag = cidx >> 6;                    // 0..23
            const int offb = (cidx & 63) << 4;             // byte offset in frag
            const int f2   = frag - 16;
            const size_t gfrag = (frag < 16)
                ? (((size_t)(i0 + (frag >> 1)) * 2 + (frag & 1)) * KSTEPS + k)
                : (((size_t)(j0 + (f2 >> 1)) * 2 + (f2 & 1)) * KSTEPS + k);
            const unsigned char* g =
                (const unsigned char*)((frag < 16) ? a_frag : b_frag) + gfrag * 1024 + offb;
            const unsigned lds = lbase + (unsigned)(frag * 1024 + offb);
            asm volatile("global_load_async_to_lds_b128 %0, %1, off"
                         :: "v"(lds), "v"(g) : "memory");
        }
    };

    // ---- Stage 2: O = A^T B for this wave's 2 pairs, K = 256 ----
    v8f acc[2][2][2] = {};                      // [jj][cm][dm]
    stage_k(0, 0);
    #pragma unroll
    for (int k = 0; k < KSTEPS; ++k) {
        if (k + 1 < KSTEPS) {
            stage_k(k + 1, (k + 1) & 1);        // prefetch next k-step (other buffer)
            asm volatile("s_wait_asynccnt 0x3" ::: "memory");   // k's 3 chunks landed
        } else {
            asm volatile("s_wait_asynccnt 0x0" ::: "memory");
        }
        __syncthreads();                        // all waves' chunks visible

        const __bf16* sb = &stagebuf[k & 1][0];
        v16bf A0 = *(const v16bf*)(sb + (il * 2 + 0) * FRAG + lane * 16);
        v16bf A1 = *(const v16bf*)(sb + (il * 2 + 1) * FRAG + lane * 16);
        v16bf B[2][2];
        #pragma unroll
        for (int jj = 0; jj < 2; ++jj)
            #pragma unroll
            for (int dm = 0; dm < 2; ++dm)
                B[jj][dm] = *(const v16bf*)(sb + (16 + (jl0 + jj) * 2 + dm) * FRAG + lane * 16);

        #pragma unroll
        for (int jj = 0; jj < 2; ++jj) {
            acc[jj][0][0] = WMMA_BF16(A0, B[jj][0], acc[jj][0][0]);
            acc[jj][0][1] = WMMA_BF16(A0, B[jj][1], acc[jj][0][1]);
            acc[jj][1][0] = WMMA_BF16(A1, B[jj][0], acc[jj][1][0]);
            acc[jj][1][1] = WMMA_BF16(A1, B[jj][1], acc[jj][1][1]);
        }
        __syncthreads();                        // buffer may be overwritten next iter
    }

    // ---- Scatter O (scaled) to LDS as stage-3 A-fragments.
    // C/D element (lane, r): c = cm*16 + r + 8*(lane>>4), d = dm*16 + (lane&15)
    // vec index K = c*32 + d -> kstep = c, lane' = (p&15) + 16*((d>>3)&1),
    //                           e' = (d&7) + 8*dm
    {
        const float sc = 1.0f / (float)S_DEPTH;
        #pragma unroll
        for (int jj = 0; jj < 2; ++jj) {
            const int p     = il * 4 + jl0 + jj;             // pair 0..31
            const int mt    = p >> 4;
            const int laneP = (p & 15) + (((lane >> 3) & 1) << 4);
            #pragma unroll
            for (int cm = 0; cm < 2; ++cm)
                #pragma unroll
                for (int dm = 0; dm < 2; ++dm) {
                    const int ep = (lane & 7) + (dm << 3);
                    #pragma unroll
                    for (int r = 0; r < 8; ++r) {
                        const int c = cm * 16 + r + (lane_hi << 3);   // kstep 0..31
                        ofrag[mt][c * FRAG + laneP * 16 + ep] =
                            (__bf16)(acc[jj][cm][dm][r] * sc);
                    }
                }
        }
    }
    __syncthreads();

    // ---- Stage 3: one 16x16 output tile per wave, full K = 1024 ----
    const int mt3 = w >> 3;                      // m-tile 0..1
    const int nt  = w & 7;                       // n-tile 0..7
    v8f zacc = {};
    const __bf16* op = &ofrag[mt3][0] + lane * 16;
    const __bf16* wp = w3f + (((size_t)nt * 32) << 9) + lane * 16;
    #pragma unroll
    for (int ks = 0; ks < 32; ++ks) {
        v16bf A  = *(const v16bf*)(op + ks * FRAG);
        v16bf Bf = *(const v16bf*)(wp + (size_t)ks * FRAG);
        zacc = WMMA_BF16(A, Bf, zacc);
    }

    const int och  = nt * 16 + (lane & 15);
    const float bias = b3[och];
    #pragma unroll
    for (int r = 0; r < 8; ++r) {
        const int p  = mt3 * 16 + r + (lane_hi << 3);   // pair 0..31
        const int zi = i0 + (p >> 2);
        const int zj = j0 + (p & 3);
        z[((size_t)zi * NRES + zj) * CZ + och] = zacc[r] + bias;
    }
}

// ---------------------------------------------------------------------------
extern "C" void kernel_launch(void* const* d_in, const int* in_sizes, int n_in,
                              void* d_out, int out_size, void* d_ws, size_t ws_size,
                              hipStream_t stream) {
    const float* m     = (const float*)d_in[0];
    const float* gamma = (const float*)d_in[1];
    const float* beta  = (const float*)d_in[2];
    const float* W1    = (const float*)d_in[3];
    const float* W2    = (const float*)d_in[4];
    const float* W3    = (const float*)d_in[5];
    const float* b3    = (const float*)d_in[6];
    float* z = (float*)d_out;

    // workspace carve-up (bf16 elements)
    const size_t a_elems = (size_t)NRES * 2 * KSTEPS * FRAG;   // 3,145,728
    __bf16* a_frag = (__bf16*)d_ws;
    __bf16* b_frag = a_frag + a_elems;
    __bf16* w3f    = b_frag + a_elems;                         // 8*32*512 elems

    // Stage 1: S*N waves, 8 waves per 256-thread block
    const int waves1 = S_DEPTH * NRES;
    opm_prep_ab<<<waves1 / 8, 256, 0, stream>>>(m, gamma, beta, W1, W2, a_frag, b_frag);

    // Stage 1b: 256 fragments, 8 waves per block
    opm_prep_w3<<<32, 256, 0, stream>>>(W3, w3f);

    // Stage 2+3 fused: 8x4 (i,j) tile per 512-thread workgroup
    dim3 grid(NRES / 8, NRES / 4);
    opm_main<<<grid, 512, 0, stream>>>(a_frag, b_frag, w3f, b3, z);
}
